// Olmo2Attention_44590350467845
// MI455X (gfx1250) — matrix operand order repaired and checked by reading, not compiled, not run on hardware
//
#include <hip/hip_runtime.h>
#include <hip/hip_bf16.h>

// ---------------------------------------------------------------------------
// Olmo2 attention block on gfx1250 (MI455X).
//   - all matmuls on v_wmma_f32_16x16x32_bf16 (bf16 in, fp32 accumulate)
//   - GEMM tiles staged via GLOBAL_LOAD_ASYNC_TO_LDS_B128 (ASYNCcnt DMA),
//     double-buffered LDS, s_wait_asynccnt pipelining
//   - flash attention for the score/PV stage (scores never touch HBM)
// ---------------------------------------------------------------------------

#define T_DIM 2048
#define H_DIM 2048
#define NHEAD 16
#define HD    128
#define QKV_N 6144
#define SCALE 0.08838834764831845f     // HD^-0.5
#define LN_THETA_OVER_64 0.2050369240f // ln(500000)/64

typedef __attribute__((ext_vector_type(16))) __bf16 v16bf;
typedef __attribute__((ext_vector_type(8)))  __bf16 v8bf;
typedef __attribute__((ext_vector_type(4)))  __bf16 v4bf;
typedef __attribute__((ext_vector_type(8)))  float  v8f;

__device__ __forceinline__ v8f wmma_bf16(v16bf a, v16bf b, v8f c) {
  // D = A(16x32) * B(32x16) + C(16x16), fp32 accumulate
  return __builtin_amdgcn_wmma_f32_16x16x32_bf16(
      /*neg_a=*/false, a, /*neg_b=*/false, b,
      /*c_mod=*/(short)0, c, /*reuse_a=*/false, /*reuse_b=*/false);
}

// Low 32 bits of a generic pointer to LDS == LDS byte offset (flat LDS
// addresses carry the aperture in the high DWORD; hardware truncates).
__device__ __forceinline__ unsigned lds_off(const void* p) {
  return (unsigned)(size_t)p;
}

// Async DMA: 16 bytes per lane, global -> LDS, tracked by ASYNCcnt.
__device__ __forceinline__ void async_copy_b128(unsigned lds_byte, const void* g) {
  asm volatile("global_load_async_to_lds_b128 %0, %1, off"
               :: "v"(lds_byte), "v"((unsigned long long)(size_t)g)
               : "memory");
}
__device__ __forceinline__ void wait_async_le4() {
  asm volatile("s_wait_asynccnt 0x4" ::: "memory");
}
__device__ __forceinline__ void wait_async_0() {
  asm volatile("s_wait_asynccnt 0x0" ::: "memory");
}

// Build a 16-element A-fragment from its two contiguous 8-element halves
// (lane half hl: K{0..7}+8hl and K{16..23}+8hl of the 32-wide tile row).
__device__ __forceinline__ v16bf make_afrag(const __bf16* row, int hl) {
  const v8bf lo = *(const v8bf*)(row + hl * 8);
  const v8bf hi = *(const v8bf*)(row + 16 + hl * 8);
  return __builtin_shufflevector(lo, hi, 0, 1, 2, 3, 4, 5, 6, 7,
                                 8, 9, 10, 11, 12, 13, 14, 15);
}

// ---------------------------------------------------------------------------
// One-time fp32 -> bf16 converters (streaming, trivially amortized).
// ---------------------------------------------------------------------------
__global__ __launch_bounds__(256) void cvt_bf16_kernel(
    const float* __restrict__ src, __bf16* __restrict__ dst) {
  const size_t i = (size_t)blockIdx.x * 256 + threadIdx.x;
  const float4 v = ((const float4*)src)[i];
  v4bf o = {(__bf16)v.x, (__bf16)v.y, (__bf16)v.z, (__bf16)v.w};
  *(v4bf*)(dst + i * 4) = o;
}

// src (R x C, f32 row-major) -> dst (C x R, bf16 row-major) i.e. B^T.
__global__ __launch_bounds__(256) void transpose_cvt_bf16(
    const float* __restrict__ src, __bf16* __restrict__ dst, int R, int C) {
  __shared__ __bf16 tile[32][33];
  const int r0 = blockIdx.x * 32, c0 = blockIdx.y * 32;
  const int tx = threadIdx.x & 31, ty = threadIdx.x >> 5;
  #pragma unroll
  for (int i = 0; i < 4; ++i) {
    const int r = ty + i * 8;
    tile[r][tx] = (__bf16)src[(size_t)(r0 + r) * C + c0 + tx];
  }
  __syncthreads();
  #pragma unroll
  for (int i = 0; i < 4; ++i) {
    const int r = ty + i * 8;
    dst[(size_t)(c0 + r) * R + r0 + tx] = tile[tx][r];
  }
}

// ---------------------------------------------------------------------------
// C(MxN,f32) = A(MxK,bf16 row-major) * Bt(NxK,bf16 row-major)^T via WMMA.
// Block: 256 threads = 8 waves (2x4); block tile 128x128; wave tile 64x32.
// Tiles DMA'd with async-to-LDS, double buffered on ASYNCcnt.
// ---------------------------------------------------------------------------
__global__ __launch_bounds__(256) void gemm_wmma_async(
    const __bf16* __restrict__ A, const __bf16* __restrict__ Bt,
    float* __restrict__ C, int M, int N, int K) {
  (void)M;
  __shared__ alignas(32) __bf16 As[2][128][32];
  __shared__ alignas(32) __bf16 Bs[2][128][32];

  const int tid  = threadIdx.x;
  const int lane = tid & 31;
  const int wave = tid >> 5;
  const int wm   = wave >> 2;        // 0..1 (64-row half)
  const int wn   = wave & 3;         // 0..3 (32-col quarter)
  const int hl   = lane >> 4;
  const int lc   = lane & 15;
  const int m_blk = blockIdx.x * 128;
  const int n_blk = blockIdx.y * 128;

  // copy slot: each thread moves 4x16B per tile pair (2 A rows + 2 B rows)
  const int crow = tid >> 2;            // 0..63
  const int cchk = (tid & 3) * 8;       // element offset of 16B chunk
  const __bf16* Ab = A + (size_t)m_blk * K;
  const __bf16* Bb = Bt + (size_t)n_blk * K;

  auto issue = [&](int buf, int k0) {
    async_copy_b128(lds_off(&As[buf][crow][cchk]),      Ab + (size_t)crow * K + k0 + cchk);
    async_copy_b128(lds_off(&As[buf][crow + 64][cchk]), Ab + (size_t)(crow + 64) * K + k0 + cchk);
    async_copy_b128(lds_off(&Bs[buf][crow][cchk]),      Bb + (size_t)crow * K + k0 + cchk);
    async_copy_b128(lds_off(&Bs[buf][crow + 64][cchk]), Bb + (size_t)(crow + 64) * K + k0 + cchk);
  };

  v8f acc[4][2] = {};
  issue(0, 0);
  int buf = 0;
  for (int k0 = 0; k0 < K; k0 += 32) {
    if (k0 + 32 < K) {
      issue(buf ^ 1, k0 + 32);  // prefetch next tile (4 async ops in flight)
      wait_async_le4();         // current tile's 4 done (in-order completion)
    } else {
      wait_async_0();
    }
    __syncthreads();

    v16bf afr[4], bfr[2];
    #pragma unroll
    for (int mt = 0; mt < 4; ++mt)
      afr[mt] = make_afrag(&As[buf][wm * 64 + mt * 16 + lc][0], hl);
    #pragma unroll
    for (int nt = 0; nt < 2; ++nt)
      bfr[nt] = make_afrag(&Bs[buf][wn * 32 + nt * 16 + lc][0], hl);

    #pragma unroll
    for (int mt = 0; mt < 4; ++mt)
      #pragma unroll
      for (int nt = 0; nt < 2; ++nt)
        acc[mt][nt] = wmma_bf16(afr[mt], bfr[nt], acc[mt][nt]);
    __syncthreads();
    buf ^= 1;
  }

  // C/D layout: VGPR r -> M = r + 8*lane_half; N = lane&15
  #pragma unroll
  for (int mt = 0; mt < 4; ++mt)
    #pragma unroll
    for (int nt = 0; nt < 2; ++nt)
      #pragma unroll
      for (int r = 0; r < 8; ++r) {
        const int row = m_blk + wm * 64 + mt * 16 + r + hl * 8;
        const int col = n_blk + wn * 32 + nt * 16 + lc;
        C[(size_t)row * N + col] = acc[mt][nt][r];
      }
}

// ---------------------------------------------------------------------------
// Per-token RMSNorm (over full H) + RoPE; emits bf16 Q,K head-major (h,t,d)
// and V transposed (h,d,t) for contiguous PV B-fragments.
// ---------------------------------------------------------------------------
__global__ __launch_bounds__(128) void norm_rope_kernel(
    const float* __restrict__ qkv, const float* __restrict__ qw,
    const float* __restrict__ kw, const int* __restrict__ positions,
    __bf16* __restrict__ qh, __bf16* __restrict__ kh,
    __bf16* __restrict__ vt) {
  const int t = blockIdx.x;
  const float* qrow = qkv + (size_t)t * QKV_N;
  const float* krow = qrow + H_DIM;
  const float* vrow = qrow + 2 * H_DIM;

  float sq = 0.f, sk = 0.f;
  for (int i = threadIdx.x; i < H_DIM; i += 128) {
    const float a = qrow[i]; sq += a * a;
    const float b = krow[i]; sk += b * b;
  }
  #pragma unroll
  for (int off = 16; off; off >>= 1) {
    sq += __shfl_xor(sq, off, 32);
    sk += __shfl_xor(sk, off, 32);
  }
  __shared__ float rq[4], rk[4];
  if ((threadIdx.x & 31) == 0) {
    rq[threadIdx.x >> 5] = sq;
    rk[threadIdx.x >> 5] = sk;
  }
  __syncthreads();
  const float invq = rsqrtf((rq[0] + rq[1] + rq[2] + rq[3]) * (1.f / H_DIM) + 1e-6f);
  const float invk = rsqrtf((rk[0] + rk[1] + rk[2] + rk[3]) * (1.f / H_DIM) + 1e-6f);
  const float pos  = (float)positions[t];

  for (int idx = threadIdx.x; idx < NHEAD * (HD / 2); idx += 128) {
    const int h = idx >> 6;
    const int j = idx & 63;
    const float fr = pos * __expf(-(float)j * LN_THETA_OVER_64);
    float s, c;
    __sincosf(fr, &s, &c);
    const int base = h * HD + j;
    {
      const float x1 = qrow[base] * invq * qw[base];
      const float x2 = qrow[base + 64] * invq * qw[base + 64];
      __bf16* o = qh + ((size_t)h * T_DIM + t) * HD + j;
      o[0]  = (__bf16)(x1 * c - x2 * s);
      o[64] = (__bf16)(x2 * c + x1 * s);
    }
    {
      const float x1 = krow[base] * invk * kw[base];
      const float x2 = krow[base + 64] * invk * kw[base + 64];
      __bf16* o = kh + ((size_t)h * T_DIM + t) * HD + j;
      o[0]  = (__bf16)(x1 * c - x2 * s);
      o[64] = (__bf16)(x2 * c + x1 * s);
    }
  }
  for (int idx = threadIdx.x; idx < H_DIM; idx += 128) {
    const int h = idx >> 7;
    const int d = idx & 127;
    vt[((size_t)h * HD + d) * T_DIM + t] = (__bf16)vrow[idx];
  }
}

// ---------------------------------------------------------------------------
// Flash attention: one wave per (head, 16-query tile); key blocks of 32.
// Scores: 8 WMMAs/block; online softmax with 16-lane row reductions;
// probs staged through per-wave LDS (contiguous A order); PV: 8 WMMAs/block.
// ---------------------------------------------------------------------------
__global__ __launch_bounds__(128) void attn_fa_kernel(
    const __bf16* __restrict__ qh, const __bf16* __restrict__ kh,
    const __bf16* __restrict__ vt, __bf16* __restrict__ attnb) {
  const int wave = threadIdx.x >> 5;
  const int lane = threadIdx.x & 31;
  const int hl   = lane >> 4;
  const int lc   = lane & 15;
  const int gw   = blockIdx.x * 4 + wave;   // 0..2047
  const int h    = gw >> 7;                 // T/16 = 128 q-tiles per head
  const int q0   = (gw & 127) * 16;

  __shared__ alignas(32) __bf16 Pl[4][16][32];
  __bf16 (*P)[32] = Pl[wave];

  const __bf16* Q  = qh + (size_t)h * T_DIM * HD;
  const __bf16* Kp = kh + (size_t)h * T_DIM * HD;
  const __bf16* Vp = vt + (size_t)h * HD * T_DIM;

  // Q A-fragments for all 4 head-dim chunks of K=32 (held in VGPRs all loop)
  v16bf qf[4];
  {
    const __bf16* qr = Q + (size_t)(q0 + lc) * HD;
    #pragma unroll
    for (int c = 0; c < 4; ++c)
      #pragma unroll
      for (int v = 0; v < 8; ++v) {
        const int k = c * 32 + 2 * v + hl * 16;
        qf[c][2 * v]     = qr[k];
        qf[c][2 * v + 1] = qr[k + 1];
      }
  }

  v8f Oc[8] = {};
  float mrow[8], lrow[8];
  #pragma unroll
  for (int r = 0; r < 8; ++r) { mrow[r] = -1e30f; lrow[r] = 0.f; }

  for (int kb = 0; kb < q0 + 16; kb += 32) {
    // ---- scores: two 16-key tiles, K-dim = 128 head dim in 4 chunks ----
    v8f s0 = {}, s1 = {};
    #pragma unroll
    for (int c = 0; c < 4; ++c) {
      const v16bf kf0 = *(const v16bf*)(Kp + (size_t)(kb + lc) * HD + c * 32 + hl * 16);
      const v16bf kf1 = *(const v16bf*)(Kp + (size_t)(kb + 16 + lc) * HD + c * 32 + hl * 16);
      s0 = wmma_bf16(qf[c], kf0, s0);
      s1 = wmma_bf16(qf[c], kf1, s1);
    }
    // ---- causal mask + online softmax (rows live in half-wave groups) ----
    const int qrow0 = q0 + hl * 8;
    const int kc0 = kb + lc, kc1 = kb + 16 + lc;
    #pragma unroll
    for (int r = 0; r < 8; ++r) {
      const int q = qrow0 + r;
      const float x0 = (kc0 <= q) ? s0[r] * SCALE : -1e30f;
      const float x1 = (kc1 <= q) ? s1[r] * SCALE : -1e30f;
      float mloc = fmaxf(x0, x1);
      #pragma unroll
      for (int off = 8; off; off >>= 1)
        mloc = fmaxf(mloc, __shfl_xor(mloc, off, 16));
      const float mnew  = fmaxf(mrow[r], mloc);
      const float alpha = __expf(mrow[r] - mnew);
      const float p0 = __expf(x0 - mnew);
      const float p1 = __expf(x1 - mnew);
      float ls = p0 + p1;
      #pragma unroll
      for (int off = 8; off; off >>= 1) ls += __shfl_xor(ls, off, 16);
      lrow[r] = lrow[r] * alpha + ls;
      mrow[r] = mnew;
      #pragma unroll
      for (int dt = 0; dt < 8; ++dt) Oc[dt][r] *= alpha;
      // stage probs in contiguous A order: k=lc -> slot lc; k=16+lc -> slot 16+lc
      const int prow = r + hl * 8;
      P[prow][lc]      = (__bf16)p0;
      P[prow][16 + lc] = (__bf16)p1;
    }
    asm volatile("" ::: "memory");  // LDS unit is in-order per wave; fence the compiler
    const v16bf pa = *(const v16bf*)&P[lc][hl * 16];
    // ---- PV: V^T layout (h,d,t) makes B-fragments contiguous ----
    #pragma unroll
    for (int dt = 0; dt < 8; ++dt) {
      const v16bf vf =
          *(const v16bf*)(Vp + (size_t)(dt * 16 + lc) * T_DIM + kb + hl * 16);
      Oc[dt] = wmma_bf16(pa, vf, Oc[dt]);
    }
    asm volatile("" ::: "memory");
  }

  #pragma unroll
  for (int dt = 0; dt < 8; ++dt)
    #pragma unroll
    for (int r = 0; r < 8; ++r) {
      const int row = q0 + r + hl * 8;
      const int col = h * HD + dt * 16 + lc;
      attnb[(size_t)row * H_DIM + col] = (__bf16)(Oc[dt][r] / lrow[r]);
    }
}

// ---------------------------------------------------------------------------
extern "C" void kernel_launch(void* const* d_in, const int* in_sizes, int n_in,
                              void* d_out, int out_size, void* d_ws, size_t ws_size,
                              hipStream_t stream) {
  (void)in_sizes; (void)n_in; (void)out_size; (void)ws_size;
  const int*   positions = (const int*)  d_in[0];
  const float* hidden    = (const float*)d_in[1];
  const float* w_qkv     = (const float*)d_in[2];
  const float* q_norm_w  = (const float*)d_in[3];
  const float* k_norm_w  = (const float*)d_in[4];
  const float* w_o       = (const float*)d_in[5];
  float* out = (float*)d_out;

  // Workspace layout (112 MB):
  //  [0,48M)   qkv fp32 (GEMM1 out, consumed by norm_rope)
  //  [0,8M)    attnb bf16 ALIASES qkv region (written only after norm_rope)
  //  [48,56M)  hidden bf16
  //  [56,80M)  w_qkv^T bf16 (N x K)
  //  [80,88M)  w_o^T  bf16 (N x K)
  //  [88..112M) q (h,t,d) / k (h,t,d) / v^T (h,d,t) bf16
  char* ws = (char*)d_ws;
  float*  qkv   = (float*)ws;
  __bf16* attnb = (__bf16*)ws;  // safe alias: lifetime disjoint from qkv
  __bf16* hidb  = (__bf16*)(ws + (size_t)T_DIM * QKV_N * 4);
  __bf16* wqkvt = hidb  + (size_t)T_DIM * H_DIM;
  __bf16* wot   = wqkvt + (size_t)QKV_N * H_DIM;
  __bf16* qh    = wot   + (size_t)H_DIM * H_DIM;
  __bf16* khb   = qh    + (size_t)NHEAD * T_DIM * HD;
  __bf16* vtb   = khb   + (size_t)NHEAD * T_DIM * HD;

  // 0) one-time bf16 conversions (+ weight transposes for [N][K] B layout)
  cvt_bf16_kernel<<<(T_DIM * H_DIM / 4) / 256, 256, 0, stream>>>(hidden, hidb);
  transpose_cvt_bf16<<<dim3(H_DIM / 32, QKV_N / 32), 256, 0, stream>>>(
      w_qkv, wqkvt, H_DIM, QKV_N);
  transpose_cvt_bf16<<<dim3(H_DIM / 32, H_DIM / 32), 256, 0, stream>>>(
      w_o, wot, H_DIM, H_DIM);
  // 1) qkv = hidden @ w_qkv   (2048x2048x6144)
  gemm_wmma_async<<<dim3(T_DIM / 128, QKV_N / 128), 256, 0, stream>>>(
      hidb, wqkvt, qkv, T_DIM, QKV_N, H_DIM);
  // 2) RMSNorm + RoPE -> bf16 Q,K (h,t,d) and V^T (h,d,t)
  norm_rope_kernel<<<T_DIM, 128, 0, stream>>>(
      qkv, q_norm_w, k_norm_w, positions, qh, khb, vtb);
  // 3) flash attention -> attn (T, H) bf16
  attn_fa_kernel<<<(NHEAD * T_DIM / 16) / 4, 128, 0, stream>>>(
      qh, khb, vtb, attnb);
  // 4) out = attn @ w_o   (2048x2048x2048)
  gemm_wmma_async<<<dim3(T_DIM / 128, H_DIM / 128), 256, 0, stream>>>(
      attnb, wot, out, T_DIM, H_DIM, H_DIM);
}
